// LSTM_90572270338763
// MI455X (gfx1250) — matrix operand order repaired and checked by reading, not compiled, not run on hardware
//
#include <hip/hip_runtime.h>
#include <math.h>

// Problem constants (from reference): B=512, T=40, C=512, H=512
#define BB 512
#define TT 40
#define CC 512
#define HH 512
#define G3 1536            // 3*H
#define MROWS (BB * TT)    // 20480
#define NBLK (G3 / 32)     // 48  (32-wide N blocks, 2x2 tile blocking)

typedef __bf16 bf16_t;
typedef __attribute__((ext_vector_type(16))) __bf16 v16bf;
typedef __attribute__((ext_vector_type(8)))  float  v8f;

union ABFrag { uint4 u[2]; v16bf v; };

__device__ __forceinline__ unsigned short f2bf(float f) {
    unsigned int u = __float_as_uint(f);
    u = (u + 0x7fffu + ((u >> 16) & 1u)) >> 16;   // round-to-nearest-even
    return (unsigned short)u;
}

// A fragment: 16x32 (MxK) bf16, row-major source with leading dim ld.
// ISA layout (16-bit A 16x32): lane L holds row M=L%16;
//   lanes 0-15:  K = kb+0..7 (v0..v3), kb+16..23 (v4..v7)
//   lanes 16-31: K = kb+8..15,         kb+24..31
__device__ __forceinline__ v16bf load_a_frag(const unsigned short* __restrict__ X,
                                             int ld, int m0, int kb, int lane) {
    const int m = lane & 15, half = lane >> 4;
    const unsigned short* p = X + (size_t)(m0 + m) * ld + kb + half * 8;
    ABFrag f;
    f.u[0] = *reinterpret_cast<const uint4*>(p);
    f.u[1] = *reinterpret_cast<const uint4*>(p + 16);
    return f.v;
}

// B fragment: 32x16 (KxN) bf16. Weight stored row-major [N, K] (PyTorch W[g,c]),
// so column n of B = row n of W -> contiguous K. Lanes 0-15 hold K=kb+0..7, kb+16..23
// for col N=lane; lanes 16-31 hold K=kb+8..15, kb+24..31 for col N=lane-16.
__device__ __forceinline__ v16bf load_b_frag(const unsigned short* __restrict__ W,
                                             int ld, int n0, int kb, int lane) {
    const int n = lane & 15, khalf = lane >> 4;
    const unsigned short* p = W + (size_t)(n0 + n) * ld + kb + khalf * 8;
    ABFrag f;
    f.u[0] = *reinterpret_cast<const uint4*>(p);
    f.u[1] = *reinterpret_cast<const uint4*>(p + 16);
    return f.v;
}

// Store one 16x16 f32 D tile (+optional per-column bias add done by caller).
__device__ __forceinline__ void store_d(float* __restrict__ out, int ldo,
                                        int m0, int n0, int lane,
                                        const v8f& c, float bias) {
    const int col = lane & 15, half = lane >> 4;
#pragma unroll
    for (int j = 0; j < 8; ++j) {
        int row = m0 + j + 8 * half;   // D layout: VGPR j -> M = j + 8*(lane/16)
        out[(size_t)row * ldo + n0 + col] = c[j] + bias;
    }
}

// ---------------- conversion / init / mask kernels ----------------

__global__ void k_convert(const float* __restrict__ in, unsigned short* __restrict__ out, int n) {
    int i = blockIdx.x * blockDim.x + threadIdx.x;
    if (i < n) out[i] = f2bf(in[i]);
}

__global__ void k_valid(const float* __restrict__ x, int* __restrict__ valid) {
    int i = blockIdx.x * blockDim.x + threadIdx.x;   // i = b*T + t
    if (i >= BB * TT) return;
    const float* row = x + (size_t)i * CC;
    int v = 0;
    for (int c = 0; c < CC; ++c) v |= (row[c] != 0.0f);
    valid[i] = v;
}

__global__ void k_init(float* __restrict__ out_max, float* __restrict__ h,
                       unsigned short* __restrict__ hbf) {
    int i = blockIdx.x * blockDim.x + threadIdx.x;   // n = 2*B*H = 524288
    if (i >= 2 * BB * HH) return;
    out_max[i] = -3.402823466e38f;
    h[i] = 0.0f;
    hbf[i] = 0;
}

// ------- GEMM: gx = x @ Wih^T + bih  (M=20480, N=1536, K=512), 2x2 blocked ---

__global__ void __launch_bounds__(256) k_gemm_gx(const unsigned short* __restrict__ Xbf,
                                                 const unsigned short* __restrict__ Wbf,
                                                 const float* __restrict__ bih,
                                                 float* __restrict__ gx) {
    const int lane = threadIdx.x & 31;
    const int widx = blockIdx.x * (blockDim.x >> 5) + (threadIdx.x >> 5);
    const int mtb = widx / NBLK, ntb = widx % NBLK;
    const int m0 = mtb * 32, n0 = ntb * 32;
    v8f c00 = {}, c01 = {}, c10 = {}, c11 = {};
#pragma unroll
    for (int ks = 0; ks < CC / 32; ++ks) {
        const int kb = ks * 32;
        v16bf a0 = load_a_frag(Xbf, CC, m0,      kb, lane);
        v16bf a1 = load_a_frag(Xbf, CC, m0 + 16, kb, lane);
        v16bf b0 = load_b_frag(Wbf, CC, n0,      kb, lane);
        v16bf b1 = load_b_frag(Wbf, CC, n0 + 16, kb, lane);
        c00 = __builtin_amdgcn_wmma_f32_16x16x32_bf16(false, a0, false, b0, (short)0, c00, false, false);
        c01 = __builtin_amdgcn_wmma_f32_16x16x32_bf16(false, a0, false, b1, (short)0, c01, false, false);
        c10 = __builtin_amdgcn_wmma_f32_16x16x32_bf16(false, a1, false, b0, (short)0, c10, false, false);
        c11 = __builtin_amdgcn_wmma_f32_16x16x32_bf16(false, a1, false, b1, (short)0, c11, false, false);
    }
    const int col = lane & 15;
    const float bias0 = bih[n0 + col];
    const float bias1 = bih[n0 + 16 + col];
    store_d(gx, G3, m0,      n0,      lane, c00, bias0);
    store_d(gx, G3, m0,      n0 + 16, lane, c01, bias1);
    store_d(gx, G3, m0 + 16, n0,      lane, c10, bias0);
    store_d(gx, G3, m0 + 16, n0 + 16, lane, c11, bias1);
}

// ------- GEMM: gh = h @ Whh^T  (M=1024 fwd+bwd stacked, N=1536, K=512) -------

__global__ void __launch_bounds__(256) k_gemm_gh(const unsigned short* __restrict__ hbf,
                                                 const unsigned short* __restrict__ Whf,
                                                 const unsigned short* __restrict__ Whb,
                                                 float* __restrict__ gh) {
    const int lane = threadIdx.x & 31;
    const int widx = blockIdx.x * (blockDim.x >> 5) + (threadIdx.x >> 5);
    const int mtb = widx / NBLK, ntb = widx % NBLK;
    const int m0 = mtb * 32, n0 = ntb * 32;
    const unsigned short* W = (m0 < BB) ? Whf : Whb;   // 32-row blocks never straddle 512
    v8f c00 = {}, c01 = {}, c10 = {}, c11 = {};
#pragma unroll
    for (int ks = 0; ks < HH / 32; ++ks) {
        const int kb = ks * 32;
        v16bf a0 = load_a_frag(hbf, HH, m0,      kb, lane);
        v16bf a1 = load_a_frag(hbf, HH, m0 + 16, kb, lane);
        v16bf b0 = load_b_frag(W, HH, n0,      kb, lane);
        v16bf b1 = load_b_frag(W, HH, n0 + 16, kb, lane);
        c00 = __builtin_amdgcn_wmma_f32_16x16x32_bf16(false, a0, false, b0, (short)0, c00, false, false);
        c01 = __builtin_amdgcn_wmma_f32_16x16x32_bf16(false, a0, false, b1, (short)0, c01, false, false);
        c10 = __builtin_amdgcn_wmma_f32_16x16x32_bf16(false, a1, false, b0, (short)0, c10, false, false);
        c11 = __builtin_amdgcn_wmma_f32_16x16x32_bf16(false, a1, false, b1, (short)0, c11, false, false);
    }
    store_d(gh, G3, m0,      n0,      lane, c00, 0.0f);
    store_d(gh, G3, m0,      n0 + 16, lane, c01, 0.0f);
    store_d(gh, G3, m0 + 16, n0,      lane, c10, 0.0f);
    store_d(gh, G3, m0 + 16, n0 + 16, lane, c11, 0.0f);
}

// ---------------- GRU gate update + masked running max -----------------------

__global__ void __launch_bounds__(256) k_gate(int t,
                                              const float* __restrict__ gh,
                                              const float* __restrict__ gx_f,
                                              const float* __restrict__ gx_b,
                                              const float* __restrict__ bhh_f,
                                              const float* __restrict__ bhh_b,
                                              float* __restrict__ h,
                                              unsigned short* __restrict__ hbf,
                                              const int* __restrict__ valid,
                                              float* __restrict__ out_max) {
    int i = blockIdx.x * blockDim.x + threadIdx.x;   // 2*B*H threads
    if (i >= 2 * BB * HH) return;
    const int j = i & (HH - 1);
    const int db = i >> 9;            // 0..1023 (dir*B + b)
    const int d = db >> 9;            // 0 fwd, 1 bwd
    const int b = db & (BB - 1);
    const int t_idx = (d == 0) ? t : (TT - 1 - t);

    const float* gx  = (d == 0) ? gx_f : gx_b;
    const float* bhh = (d == 0) ? bhh_f : bhh_b;

    const size_t gxo = ((size_t)b * TT + t_idx) * G3;
    const size_t gho = (size_t)db * G3;

    const float gr  = gx[gxo + j]            + gh[gho + j]            + bhh[j];
    const float gz  = gx[gxo + HH + j]       + gh[gho + HH + j]       + bhh[HH + j];
    const float ghn = gh[gho + 2 * HH + j]   + bhh[2 * HH + j];
    const float gxn = gx[gxo + 2 * HH + j];

    const float r = 1.0f / (1.0f + __expf(-gr));
    const float z = 1.0f / (1.0f + __expf(-gz));
    const float n = tanhf(gxn + r * ghn);
    const float hold = h[(size_t)db * HH + j];
    const float hn = (1.0f - z) * n + z * hold;

    h[(size_t)db * HH + j]   = hn;
    hbf[(size_t)db * HH + j] = f2bf(hn);

    if (valid[b * TT + t_idx]) {
        float* o = out_max + (size_t)b * (2 * HH) + d * HH + j;
        *o = fmaxf(*o, hn);
    }
}

// ---------------- host-side orchestration ------------------------------------

extern "C" void kernel_launch(void* const* d_in, const int* in_sizes, int n_in,
                              void* d_out, int out_size, void* d_ws, size_t ws_size,
                              hipStream_t stream) {
    (void)in_sizes; (void)n_in; (void)out_size; (void)ws_size;

    const float* x     = (const float*)d_in[0];   // [B,T,C]
    const float* Wih_f = (const float*)d_in[1];   // [3H,C]
    const float* Whh_f = (const float*)d_in[2];   // [3H,H]
    const float* bih_f = (const float*)d_in[3];
    const float* bhh_f = (const float*)d_in[4];
    const float* Wih_b = (const float*)d_in[5];
    const float* Whh_b = (const float*)d_in[6];
    const float* bih_b = (const float*)d_in[7];
    const float* bhh_b = (const float*)d_in[8];
    float* out = (float*)d_out;                   // [B,1,2H]

    // Workspace layout (256B aligned slices)
    char* ws = (char*)d_ws;
    size_t off = 0;
    auto take = [&](size_t bytes) { char* p = ws + off; off = (off + bytes + 255) & ~(size_t)255; return p; };

    unsigned short* x_bf   = (unsigned short*)take((size_t)MROWS * CC * 2);  // 21.0 MB
    unsigned short* Wih_fb = (unsigned short*)take((size_t)G3 * CC * 2);     // 1.57 MB
    unsigned short* Whh_fb = (unsigned short*)take((size_t)G3 * HH * 2);
    unsigned short* Wih_bb = (unsigned short*)take((size_t)G3 * CC * 2);
    unsigned short* Whh_bb = (unsigned short*)take((size_t)G3 * HH * 2);
    float* gx_f = (float*)take((size_t)MROWS * G3 * 4);                      // 125.8 MB
    float* gx_b = (float*)take((size_t)MROWS * G3 * 4);                      // 125.8 MB
    float* gh   = (float*)take((size_t)2 * BB * G3 * 4);                     // 6.3 MB
    float* h    = (float*)take((size_t)2 * BB * HH * 4);                     // 2.1 MB
    unsigned short* h_bf = (unsigned short*)take((size_t)2 * BB * HH * 2);   // 1.0 MB
    int* valid = (int*)take((size_t)BB * TT * 4);

    // 1) f32 -> bf16 conversions
    {
        int n = MROWS * CC;
        k_convert<<<(n + 255) / 256, 256, 0, stream>>>(x, x_bf, n);
        n = G3 * CC;
        k_convert<<<(n + 255) / 256, 256, 0, stream>>>(Wih_f, Wih_fb, n);
        k_convert<<<(n + 255) / 256, 256, 0, stream>>>(Wih_b, Wih_bb, n);
        n = G3 * HH;
        k_convert<<<(n + 255) / 256, 256, 0, stream>>>(Whh_f, Whh_fb, n);
        k_convert<<<(n + 255) / 256, 256, 0, stream>>>(Whh_b, Whh_bb, n);
    }

    // 2) validity mask + state / output-max init
    k_valid<<<(BB * TT + 255) / 256, 256, 0, stream>>>(x, valid);
    k_init<<<(2 * BB * HH + 255) / 256, 256, 0, stream>>>(out, h, h_bf);

    // 3) input projections for all timesteps (big WMMA GEMMs, 2x2 blocked)
    {
        const int waves = (MROWS / 32) * NBLK;         // 640 * 48 = 30720 waves
        const int blocks = waves / 8;                  // 8 waves / 256-thread block
        k_gemm_gx<<<blocks, 256, 0, stream>>>(x_bf, Wih_fb, bih_f, gx_f);
        k_gemm_gx<<<blocks, 256, 0, stream>>>(x_bf, Wih_bb, bih_b, gx_b);
    }

    // 4) sequential scan: per step one recurrent GEMM (fwd+bwd fused) + gates
    {
        const int gh_waves = (2 * BB / 32) * NBLK;     // 32 * 48 = 1536 waves
        const int gh_blocks = gh_waves / 8;            // 192 blocks
        const int gate_blocks = (2 * BB * HH) / 256;   // 2048 blocks
        for (int t = 0; t < TT; ++t) {
            k_gemm_gh<<<gh_blocks, 256, 0, stream>>>(h_bf, Whh_fb, Whh_bb, gh);
            k_gate<<<gate_blocks, 256, 0, stream>>>(t, gh, gx_f, gx_b, bhh_f, bhh_b,
                                                    h, h_bf, valid, out);
        }
    }
}
